// attn_decoder_62955630624855
// MI455X (gfx1250) — compile-verified, matrix operand order checked by reading, not run on hardware
//
#include <hip/hip_runtime.h>
#include <hip/hip_bf16.h>

// ---------------- problem constants ----------------
#define B_   64
#define L_   400
#define H_   256
#define E_   128
#define V_   50000
#define OOV_ 400
#define VT_  50400
#define K3H  768          // 3*H, K of pv GEMM
#define KENC 912          // 2H + L
#define KPG  896          // 3H + E

typedef __attribute__((ext_vector_type(16))) __bf16 v16bf;
typedef __attribute__((ext_vector_type(8)))  float  v8f;

__device__ __forceinline__ unsigned short f2bf(float f) {
    unsigned u = __float_as_uint(f);
    unsigned r = (u + 0x7FFFu + ((u >> 16) & 1u)) >> 16;   // round-to-nearest-even
    return (unsigned short)r;
}
__device__ __forceinline__ float sigm(float x) { return 1.0f / (1.0f + __expf(-x)); }

__device__ __forceinline__ float blk_reduce_max(float v, float* red) {
    int t = threadIdx.x;
    red[t] = v; __syncthreads();
    for (int s = 128; s > 0; s >>= 1) { if (t < s) red[t] = fmaxf(red[t], red[t + s]); __syncthreads(); }
    float r = red[0]; __syncthreads(); return r;
}
__device__ __forceinline__ float blk_reduce_sum(float v, float* red) {
    int t = threadIdx.x;
    red[t] = v; __syncthreads();
    for (int s = 128; s > 0; s >>= 1) { if (t < s) red[t] += red[t + s]; __syncthreads(); }
    float r = red[0]; __syncthreads(); return r;
}

// ---------------- 1) embedding gather + LSTM cell ----------------
__global__ __launch_bounds__(256) void k_lstm(
    const int* __restrict__ dec_inp, const float* __restrict__ embed,
    const float* __restrict__ dec_h, const float* __restrict__ dec_c,
    const float* __restrict__ W_ih, const float* __restrict__ W_hh,
    const float* __restrict__ b_ih, const float* __restrict__ b_hh,
    float* __restrict__ out_h1, float* __restrict__ out_c1, float* __restrict__ ws_emb)
{
    __shared__ float se[E_];
    __shared__ float sh[H_];
    __shared__ float sg[4 * H_];
    int b = blockIdx.x, t = threadIdx.x;
    int tok = dec_inp[b];
    if (t < E_) { float v = embed[(size_t)tok * E_ + t]; se[t] = v; ws_emb[b * E_ + t] = v; }
    sh[t] = dec_h[b * H_ + t];
    __syncthreads();
    #pragma unroll
    for (int jj = 0; jj < 4; ++jj) {
        int j = t + jj * 256;
        float acc = b_ih[j] + b_hh[j];
        const float* wi = W_ih + (size_t)j * E_;
        for (int k = 0; k < E_; ++k) acc += se[k] * wi[k];
        const float* wh = W_hh + (size_t)j * H_;
        for (int k = 0; k < H_; ++k) acc += sh[k] * wh[k];
        sg[j] = acc;
    }
    __syncthreads();
    float gi = sg[t], gf = sg[t + H_], gg = sg[t + 2 * H_], go = sg[t + 3 * H_];
    float cprev = dec_c[b * H_ + t];
    float c1 = sigm(gf) * cprev + sigm(gi) * tanhf(gg);
    float h1 = sigm(go) * tanhf(c1);
    out_h1[b * H_ + t] = h1;
    out_c1[b * H_ + t] = c1;
}

// ---------------- 2) attention logits + softmax + coverage + context + Xbf16 staging ----------------
__global__ __launch_bounds__(256) void k_attn(
    const float* __restrict__ h1, const float* __restrict__ c1,
    const float* __restrict__ coverage, const float* __restrict__ W_enc,
    const float* __restrict__ b_enc, const float* __restrict__ enc_output,
    float* __restrict__ out_cov, float* __restrict__ ws_attn,
    float* __restrict__ ws_ctx, unsigned short* __restrict__ ws_xbf)
{
    __shared__ float st[KENC];
    __shared__ float sa[L_];
    __shared__ float red[256];
    int b = blockIdx.x, t = threadIdx.x;
    st[t]        = h1[b * H_ + t];
    st[H_ + t]   = c1[b * H_ + t];
    for (int i = t; i < L_; i += 256) st[2 * H_ + i] = coverage[b * L_ + i];
    __syncthreads();
    for (int l = t; l < L_; l += 256) {
        float acc = b_enc[l];
        const float* w = W_enc + (size_t)l * KENC;
        for (int k = 0; k < KENC; ++k) acc += st[k] * w[k];
        sa[l] = fmaxf(acc, 0.0f);                      // relu
    }
    __syncthreads();
    float m = -3.4e38f;
    for (int l = t; l < L_; l += 256) m = fmaxf(m, sa[l]);
    m = blk_reduce_max(m, red);
    float s = 0.0f;
    for (int l = t; l < L_; l += 256) { float e = __expf(sa[l] - m); sa[l] = e; s += e; }
    s = blk_reduce_sum(s, red);
    float inv = 1.0f / s;
    for (int l = t; l < L_; l += 256) {
        float a = sa[l] * inv; sa[l] = a;
        ws_attn[b * L_ + l] = a;
        out_cov[b * L_ + l] = st[2 * H_ + l] + a;
    }
    __syncthreads();
    // context[e] = sum_l attn[l] * enc[b,l,e]  (coalesced across e)
    for (int e = t; e < 2 * H_; e += 256) {
        const float* ep = enc_output + ((size_t)b * L_) * (2 * H_) + e;
        float acc = 0.0f;
        for (int l = 0; l < L_; ++l) acc += sa[l] * ep[(size_t)l * (2 * H_)];
        ws_ctx[b * 2 * H_ + e] = acc;
        ws_xbf[(size_t)b * K3H + e] = f2bf(acc);       // X = [context | c1] in bf16
    }
    ws_xbf[(size_t)b * K3H + 2 * H_ + t] = f2bf(st[H_ + t]);   // c1 part
}

// ---------------- 3) p_gen ----------------
__global__ __launch_bounds__(256) void k_pgen(
    const float* __restrict__ ctx, const float* __restrict__ emb, const float* __restrict__ c1,
    const float* __restrict__ W_pg, const float* __restrict__ b_pg, float* __restrict__ pgen)
{
    __shared__ float red[256];
    int b = blockIdx.x, t = threadIdx.x;
    float acc = 0.0f;
    for (int i = t; i < KPG; i += 256) {
        float x = (i < 2 * H_) ? ctx[b * 2 * H_ + i]
                : (i < 2 * H_ + E_) ? emb[b * E_ + (i - 2 * H_)]
                : c1[b * H_ + (i - 2 * H_ - E_)];
        acc += x * W_pg[i];
    }
    acc = blk_reduce_sum(acc, red);
    if (t == 0) pgen[b] = sigm(acc + b_pg[0]);
}

// ---------------- 4) pv logits GEMM: (64 x 768) x (768 x 50000) via WMMA bf16 ----------------
// Block = 8 waves; wave w owns a 64x16 vocab strip. Column index is CLAMPED so the hot loop is
// branchless (EXEC all-ones, as WMMA requires); only epilogue stores are masked.
__global__ __launch_bounds__(256) void k_pv(
    const unsigned short* __restrict__ xbf, const float* __restrict__ W_pv,
    const float* __restrict__ b_pv, float* __restrict__ pf /* logits @ stride VT_ */)
{
    int t = threadIdx.x;
    int lane = t & 31, w = t >> 5;
    int vcol = blockIdx.x * 128 + w * 16 + (lane & 15);
    int vc   = vcol < V_ ? vcol : V_ - 1;      // clamp: loads stay in-bounds, loop stays convergent
    int kb   = (lane & 16) >> 1;               // lanes 0-15 -> K base 0 ; lanes 16-31 -> K base 8
    int r    = lane & 15;
    const float*          wp = W_pv + (size_t)vc * K3H + kb;
    const unsigned short* xp = xbf + (size_t)r * K3H + kb;

    v8f c0 = {}, c1v = {}, c2 = {}, c3 = {};
    #pragma unroll 2
    for (int k0 = 0; k0 < K3H; k0 += 32) {
        __builtin_prefetch(wp + k0 + 64, 0, 1);              // stream W_pv ahead (global_prefetch)
        // B fragment: column vcol, K = {k0+kb..+7} and {k0+16+kb..+7}; f32 -> bf16 in flight
        const float* p0 = wp + k0;
        v16bf bf;
        #pragma unroll
        for (int i = 0; i < 8; ++i) { bf[i] = (__bf16)p0[i]; bf[8 + i] = (__bf16)p0[16 + i]; }
        // A fragments (4 M-tiles): row m*16+r, two contiguous 16B bf16 runs each
        v16bf a0, a1, a2, a3;
        #define LOADA(am, m)                                                                  \
        {                                                                                     \
            const unsigned short* q = xp + (size_t)(m) * 16 * K3H + k0;                       \
            *(uint4*)&((unsigned short*)&(am))[0] = *(const uint4*)q;                         \
            *(uint4*)&((unsigned short*)&(am))[8] = *(const uint4*)(q + 16);                  \
        }
        LOADA(a0, 0) LOADA(a1, 1) LOADA(a2, 2) LOADA(a3, 3)
        #undef LOADA
        c0  = __builtin_amdgcn_wmma_f32_16x16x32_bf16(false, a0, false, bf, (short)0, c0,  false, false);
        c1v = __builtin_amdgcn_wmma_f32_16x16x32_bf16(false, a1, false, bf, (short)0, c1v, false, false);
        c2  = __builtin_amdgcn_wmma_f32_16x16x32_bf16(false, a2, false, bf, (short)0, c2,  false, false);
        c3  = __builtin_amdgcn_wmma_f32_16x16x32_bf16(false, a3, false, bf, (short)0, c3,  false, false);
    }
    if (vcol >= V_) return;                    // mask only the stores
    float bias = b_pv[vcol];
    int rb = (lane >> 4) << 3;                 // C/D: row = m*16 + rb + j, col = lane&15
    #pragma unroll
    for (int j = 0; j < 8; ++j) {
        pf[(size_t)(0  + rb + j) * VT_ + vcol] = c0[j]  + bias;
        pf[(size_t)(16 + rb + j) * VT_ + vcol] = c1v[j] + bias;
        pf[(size_t)(32 + rb + j) * VT_ + vcol] = c2[j]  + bias;
        pf[(size_t)(48 + rb + j) * VT_ + vcol] = c3[j]  + bias;
    }
}

// ---------------- 5) row softmax over 50000 logits (in place: pf <- exp(x - max)), rowsum saved ----------------
__global__ __launch_bounds__(256) void k_softmax(float* __restrict__ pf, float* __restrict__ rowsum)
{
    __shared__ float red[256];
    int b = blockIdx.x, t = threadIdx.x;
    float* row = pf + (size_t)b * VT_;
    float m = -3.4e38f;
    for (int v = t; v < V_; v += 256) m = fmaxf(m, row[v]);
    m = blk_reduce_max(m, red);
    float s = 0.0f;
    for (int v = t; v < V_; v += 256) { float e = __expf(row[v] - m); row[v] = e; s += e; }
    s = blk_reduce_sum(s, red);
    if (t == 0) rowsum[b] = s;
}

// ---------------- 6) token counts ----------------
__global__ __launch_bounds__(256) void k_counts(const int* __restrict__ ids, int* __restrict__ counts)
{
    int i = blockIdx.x * 256 + threadIdx.x;
    if (i < B_ * L_) atomicAdd(&counts[ids[i]], 1);
}

// ---------------- 7) S = match @ attn, attn2, p_copy scatter ----------------
__global__ __launch_bounds__(256) void k_dup(
    const int* __restrict__ ids, const float* __restrict__ attn, const int* __restrict__ counts,
    float* __restrict__ out_attn2, float* __restrict__ pcopy)
{
    __shared__ int   sid[L_];
    __shared__ float sat[L_];
    int b = blockIdx.x, t = threadIdx.x;
    for (int i = t; i < L_; i += 256) { sid[i] = ids[b * L_ + i]; sat[i] = attn[b * L_ + i]; }
    __syncthreads();
    for (int j = t; j < L_; j += 256) {
        int id = sid[j];
        float S = 0.0f;
        for (int k = 0; k < L_; ++k) S += (sid[k] == id) ? sat[k] : 0.0f;
        bool dup = (counts[id] > 1) && (id != 0);
        float a2 = dup ? sat[j] * S : sat[j];
        out_attn2[b * L_ + j] = a2;
        atomicAdd(&pcopy[(size_t)b * VT_ + id], a2);
    }
}

// ---------------- 8) final blend: p_final = pg * pv + (1-pg) * p_copy (in place over logits buf) ----------------
__global__ __launch_bounds__(256) void k_final(
    float* __restrict__ pf, const float* __restrict__ pcopy,
    const float* __restrict__ pgen, const float* __restrict__ rowsum)
{
    size_t idx = (size_t)blockIdx.x * 256 + threadIdx.x;
    if (idx >= (size_t)B_ * VT_) return;
    int b = (int)(idx / VT_);
    int v = (int)(idx - (size_t)b * VT_);
    float pg = pgen[b];
    float pv = (v < V_) ? pf[idx] * (1.0f / rowsum[b]) : 1e-12f;
    pf[idx] = pg * pv + (1.0f - pg) * pcopy[idx];
}

// ---------------- launch ----------------
extern "C" void kernel_launch(void* const* d_in, const int* in_sizes, int n_in,
                              void* d_out, int out_size, void* d_ws, size_t ws_size,
                              hipStream_t stream)
{
    const float* enc_output = (const float*)d_in[0];
    const int*   dec_inp    = (const int*)  d_in[1];
    const int*   input_ids  = (const int*)  d_in[2];
    const float* dec_h      = (const float*)d_in[3];
    const float* dec_c      = (const float*)d_in[4];
    const float* coverage   = (const float*)d_in[5];
    const float* embed      = (const float*)d_in[6];
    const float* W_ih       = (const float*)d_in[7];
    const float* W_hh       = (const float*)d_in[8];
    const float* b_ih       = (const float*)d_in[9];
    const float* b_hh       = (const float*)d_in[10];
    const float* W_enc      = (const float*)d_in[11];
    const float* b_enc      = (const float*)d_in[12];
    const float* W_pv       = (const float*)d_in[13];
    const float* b_pv       = (const float*)d_in[14];
    const float* W_pg       = (const float*)d_in[15];
    const float* b_pg       = (const float*)d_in[16];

    float* out = (float*)d_out;
    float* out_h1    = out;                            // 16384
    float* out_c1    = out + 16384;                    // 16384
    float* pf        = out + 32768;                    // 64*50400  (logits -> exp -> p_final)
    float* out_cov   = out + 3258368;                  // 25600
    float* out_attn2 = out + 3283968;                  // 25600
    float* out_pcopy = out + 3309568;                  // 64*50400

    char* ws = (char*)d_ws;
    unsigned short* xbf  = (unsigned short*)(ws);            // 64*768*2  = 98304
    float* ws_emb        = (float*)(ws + 98304);             // 64*128*4  = 32768
    float* ws_ctx        = (float*)(ws + 131072);            // 64*512*4  = 131072
    float* ws_attn       = (float*)(ws + 262144);            // 64*400*4  = 102400
    float* ws_pgen       = (float*)(ws + 364544);            // 256
    float* ws_rowsum     = (float*)(ws + 364800);            // 256
    int*   ws_counts     = (int*)  (ws + 365056);            // 50400*4   = 201600

    hipMemsetAsync(ws_counts, 0, (size_t)VT_ * sizeof(int), stream);
    hipMemsetAsync(out_pcopy, 0, (size_t)B_ * VT_ * sizeof(float), stream);

    k_lstm<<<B_, 256, 0, stream>>>(dec_inp, embed, dec_h, dec_c, W_ih, W_hh, b_ih, b_hh,
                                   out_h1, out_c1, ws_emb);
    k_attn<<<B_, 256, 0, stream>>>(out_h1, out_c1, coverage, W_enc, b_enc, enc_output,
                                   out_cov, ws_attn, ws_ctx, xbf);
    k_pgen<<<B_, 256, 0, stream>>>(ws_ctx, ws_emb, out_c1, W_pg, b_pg, ws_pgen);
    k_pv<<<(V_ + 127) / 128, 256, 0, stream>>>(xbf, W_pv, b_pv, pf);
    k_softmax<<<B_, 256, 0, stream>>>(pf, ws_rowsum);
    k_counts<<<(B_ * L_ + 255) / 256, 256, 0, stream>>>(input_ids, ws_counts);
    k_dup<<<B_, 256, 0, stream>>>(input_ids, ws_attn, ws_counts, out_attn2, out_pcopy);
    k_final<<<(B_ * VT_ + 255) / 256, 256, 0, stream>>>(pf, out_pcopy, ws_pgen, ws_rowsum);
}
